// _Seq2Seq_40243843564164
// MI455X (gfx1250) — compile-verified, matrix operand order checked by reading, not run on hardware
//
#include <hip/hip_runtime.h>
#include <hip/hip_bf16.h>

// ---------------------------------------------------------------------------
// Seq2Seq LSTM (2-layer enc, 2-layer dec) for MI455X / gfx1250.
// One workgroup per 16-row batch tile; both layers fused per timestep;
// bf16 WMMA (v_wmma_f32_16x16x32_bf16) with f32 accumulation.
// Round 4: unsigned byte offsets from a uniform base so B loads fold into
// saddr-form global_load_b128 (drops v_ashrrev/v_lshl_add_u64 address chains
// and frees ~16 VGPRs for the load scheduler).
// ---------------------------------------------------------------------------

typedef __attribute__((ext_vector_type(8)))  __bf16 v8bf;
typedef __attribute__((ext_vector_type(16))) __bf16 v16bf;
typedef __attribute__((ext_vector_type(8)))  float  v8f;

#define BB    256
#define TSRC  128
#define TTGT  32
#define HID   512
#define GG    2048        // 4*H
#define KCH   16          // K chunks of 32 (512/32)
#define NT    128         // 16-wide N tiles per 2048
#define WMAT  (NT * KCH * 32 * 16)   // elements per swizzled weight (= 2048*512)

// ---- weight pre-swizzle: fp32 (4H,H) row-major -> per-lane bf16 B-fragments.
// Fragment for (ntile, kc, lane): 16 contiguous bf16.
//   col  = ntile*16 + (lane&15)           (gate index, B-matrix column)
//   grp  = lane>>4
//   kseq = kc*32 + grp*8 + {0..7, 16..23} (matches 16-bit A/B VGPR layout)
__global__ void prep_weights(const float* __restrict__ w0, const float* __restrict__ w1,
                             const float* __restrict__ w2, const float* __restrict__ w3,
                             const float* __restrict__ w4, const float* __restrict__ w5,
                             __bf16* __restrict__ out) {
    int gidx = blockIdx.x * blockDim.x + threadIdx.x;     // 6*128*16*32 lane-frags
    if (gidx >= 6 * NT * KCH * 32) return;
    int mat   = gidx / (NT * KCH * 32);
    int rem   = gidx % (NT * KCH * 32);
    int ntile = rem / (KCH * 32);
    int r2    = rem % (KCH * 32);
    int kc    = r2 / 32;
    int lane  = r2 % 32;
    const float* w = (mat == 0) ? w0 : (mat == 1) ? w1 : (mat == 2) ? w2
                   : (mat == 3) ? w3 : (mat == 4) ? w4 : w5;
    __bf16* o = out + ((size_t)mat * (NT * KCH * 32) + (size_t)rem) * 16;
    int col = ntile * 16 + (lane & 15);
    int grp = lane >> 4;
#pragma unroll
    for (int kk = 0; kk < 16; ++kk) {
        int k = kc * 32 + grp * 8 + (kk < 8 ? kk : kk + 8);
        o[kk] = (__bf16)w[(size_t)col * HID + k];
    }
}

// Fast activations: v_exp_f32 + v_rcp_f32, no libm branches.
__device__ __forceinline__ float fast_rcp(float x) { return __builtin_amdgcn_rcpf(x); }
__device__ __forceinline__ float sigf(float x)     { return fast_rcp(1.f + __expf(-x)); }
__device__ __forceinline__ float tanh_fast(float x) {
    float e = __expf(-2.f * x);
    return (1.f - e) * fast_rcp(1.f + e);
}

// acc[8] tiles: t = q*2 + j  -> gate columns q*512 + waveId*32 + j*16 + (lane&15)
// Weight base stays uniform (SGPR pair); per-lane addressing is an unsigned
// 32-bit BYTE offset so loads fold to global_load_b128 saddr+voffset form.
__device__ __forceinline__ void gemm_acc(v8f acc[8], const __bf16* __restrict__ wfrag,
                                         const __bf16* stage, int waveId, int lane) {
    const int M   = lane & 15;
    const int grp = lane >> 4;
    const __bf16* ap = stage + M * HID + grp * 8;
    const char* wb = (const char*)wfrag;
    unsigned boff[8];
#pragma unroll
    for (int t = 0; t < 8; ++t) {
        int ntile = (t >> 1) * 32 + waveId * 2 + (t & 1);
        boff[t] = (unsigned)(((ntile * KCH) * 32 + lane) * 32);   // byte offset
    }
#pragma unroll 2
    for (int kc = 0; kc < KCH; ++kc) {
        v8bf alo = *(const v8bf*)(ap);
        v8bf ahi = *(const v8bf*)(ap + 16);
        v16bf a = __builtin_shufflevector(alo, ahi,
                     0, 1, 2, 3, 4, 5, 6, 7, 8, 9, 10, 11, 12, 13, 14, 15);
        v16bf b[8];
#pragma unroll
        for (int t = 0; t < 8; ++t) b[t] = *(const v16bf*)(wb + boff[t]);
#pragma unroll
        for (int t = 0; t < 8; ++t)
            acc[t] = __builtin_amdgcn_wmma_f32_16x16x32_bf16(
                         false, a, false, b[t], (short)0, acc[t], false, false);
#pragma unroll
        for (int t = 0; t < 8; ++t) boff[t] += 1024u;   // next K-chunk
        ap += 32;
    }
}

// gx init where input dim == 1:  acc = x[row]*w_ih[g] + bias[g]
__device__ __forceinline__ void init_gx_scalar(v8f acc[8], const float xv[8],
                                               const float* __restrict__ w_ih,
                                               const float* __restrict__ bias,
                                               int waveId, int lane) {
    const int N = lane & 15;
#pragma unroll
    for (int t = 0; t < 8; ++t) {
        int g = (t >> 1) * 512 + waveId * 32 + (t & 1) * 16 + N;
        float wv = w_ih[g];
        float bv = bias[g];
#pragma unroll
        for (int r = 0; r < 8; ++r) acc[t][r] = xv[r] * wv + bv;
    }
}

__device__ __forceinline__ void init_bias(v8f acc[8], const float* __restrict__ bias,
                                          int waveId, int lane) {
    const int N = lane & 15;
#pragma unroll
    for (int t = 0; t < 8; ++t) {
        float bv = bias[(t >> 1) * 512 + waveId * 32 + (t & 1) * 16 + N];
#pragma unroll
        for (int r = 0; r < 8; ++r) acc[t][r] = bv;
    }
}

// wave-local LSTM cell update; writes new h (bf16) into LDS stage buffer.
__device__ __forceinline__ void lstm_update(v8f acc[8], v8f cst[2], __bf16* stage,
                                            int waveId, int lane) {
    const int N     = lane & 15;
    const int rbase = (lane >> 4) * 8;
#pragma unroll
    for (int j = 0; j < 2; ++j) {
        int col = waveId * 32 + j * 16 + N;
#pragma unroll
        for (int r = 0; r < 8; ++r) {
            float iv = acc[0 + j][r], fv = acc[2 + j][r];
            float gv = acc[4 + j][r], ov = acc[6 + j][r];
            float c  = cst[j][r];
            float cn = sigf(fv) * c + sigf(iv) * tanh_fast(gv);
            float hn = sigf(ov) * tanh_fast(cn);
            cst[j][r] = cn;
            stage[(rbase + r) * HID + col] = (__bf16)hn;
        }
    }
}

__global__ __launch_bounds__(512, 1)
void seq2seq_kernel(const float* __restrict__ src,
                    const float* __restrict__ enc_w_ih0, const float* __restrict__ enc_b0,
                    const float* __restrict__ enc_b1,
                    const float* __restrict__ dec_w_ih0, const float* __restrict__ dec_b0,
                    const float* __restrict__ dec_b1,
                    const float* __restrict__ proj_w, const float* __restrict__ proj_b,
                    const __bf16* __restrict__ frag, float* __restrict__ out) {
    __shared__ __bf16 stage0[16 * HID];   // h0 (bf16 A-matrix staging)
    __shared__ __bf16 stage1[16 * HID];   // h1
    __shared__ float  last_buf[16];       // decoder scalar feedback

    const int tid    = threadIdx.x;
    const int lane   = tid & 31;
    const int waveId = tid >> 5;
    const int rbase  = (lane >> 4) * 8;
    const int b0     = blockIdx.x * 16;

    const __bf16* wEncHH0 = frag + 0 * (size_t)WMAT;
    const __bf16* wEncIH1 = frag + 1 * (size_t)WMAT;
    const __bf16* wEncHH1 = frag + 2 * (size_t)WMAT;
    const __bf16* wDecHH0 = frag + 3 * (size_t)WMAT;
    const __bf16* wDecIH1 = frag + 4 * (size_t)WMAT;
    const __bf16* wDecHH1 = frag + 5 * (size_t)WMAT;

    for (int i = tid; i < 16 * HID; i += 512) {
        stage0[i] = (__bf16)0.f;
        stage1[i] = (__bf16)0.f;
    }
    v8f c0s[2], c1s[2];
#pragma unroll
    for (int j = 0; j < 2; ++j)
#pragma unroll
        for (int r = 0; r < 8; ++r) { c0s[j][r] = 0.f; c1s[j][r] = 0.f; }
    __syncthreads();

    // ---------------- encoder: 128 fused 2-layer steps ----------------
#pragma unroll 1
    for (int t = 0; t < TSRC; ++t) {
        float xv[8];
#pragma unroll
        for (int r = 0; r < 8; ++r) xv[r] = src[(b0 + rbase + r) * TSRC + t];
        v8f acc[8];
        init_gx_scalar(acc, xv, enc_w_ih0, enc_b0, waveId, lane);
        gemm_acc(acc, wEncHH0, stage0, waveId, lane);
        __syncthreads();
        lstm_update(acc, c0s, stage0, waveId, lane);
        __syncthreads();
        init_bias(acc, enc_b1, waveId, lane);
        gemm_acc(acc, wEncIH1, stage0, waveId, lane);   // gx1 = h0_new @ W_ih1^T
        gemm_acc(acc, wEncHH1, stage1, waveId, lane);   //     + h1_old @ W_hh1^T
        __syncthreads();
        lstm_update(acc, c1s, stage1, waveId, lane);
        __syncthreads();
    }

    if (tid < 16) last_buf[tid] = src[(b0 + tid) * TSRC + (TSRC - 1)];
    __syncthreads();

    // ---------------- decoder: 32 fused 2-layer steps + projection ----------------
#pragma unroll 1
    for (int t = 0; t < TTGT; ++t) {
        float xv[8];
#pragma unroll
        for (int r = 0; r < 8; ++r) xv[r] = last_buf[rbase + r];
        v8f acc[8];
        init_gx_scalar(acc, xv, dec_w_ih0, dec_b0, waveId, lane);
        gemm_acc(acc, wDecHH0, stage0, waveId, lane);
        __syncthreads();
        lstm_update(acc, c0s, stage0, waveId, lane);
        __syncthreads();
        init_bias(acc, dec_b1, waveId, lane);
        gemm_acc(acc, wDecIH1, stage0, waveId, lane);
        gemm_acc(acc, wDecHH1, stage1, waveId, lane);
        __syncthreads();
        lstm_update(acc, c1s, stage1, waveId, lane);
        __syncthreads();
        // y = h1 . proj_w + proj_b  (one wave per batch row; wave32 reduction)
        {
            int row = tid >> 5;
            float s = 0.f;
#pragma unroll
            for (int k = lane; k < HID; k += 32)
                s += (float)stage1[row * HID + k] * proj_w[k];
#pragma unroll
            for (int off = 16; off > 0; off >>= 1) s += __shfl_down(s, off, 32);
            if (lane == 0) {
                float y = s + proj_b[0];
                last_buf[row] = y;
                out[(b0 + row) * TTGT + t] = y;
            }
        }
        __syncthreads();
    }
}

extern "C" void kernel_launch(void* const* d_in, const int* in_sizes, int n_in,
                              void* d_out, int out_size, void* d_ws, size_t ws_size,
                              hipStream_t stream) {
    const float* src       = (const float*)d_in[0];
    const float* enc_w_ih0 = (const float*)d_in[1];
    const float* enc_w_hh0 = (const float*)d_in[2];
    const float* enc_b0    = (const float*)d_in[3];
    const float* enc_w_ih1 = (const float*)d_in[4];
    const float* enc_w_hh1 = (const float*)d_in[5];
    const float* enc_b1    = (const float*)d_in[6];
    const float* dec_w_ih0 = (const float*)d_in[7];
    const float* dec_w_hh0 = (const float*)d_in[8];
    const float* dec_b0    = (const float*)d_in[9];
    const float* dec_w_ih1 = (const float*)d_in[10];
    const float* dec_w_hh1 = (const float*)d_in[11];
    const float* dec_b1    = (const float*)d_in[12];
    const float* proj_w    = (const float*)d_in[13];
    const float* proj_b    = (const float*)d_in[14];

    __bf16* frag = (__bf16*)d_ws;   // 6 * 2048*512 bf16 = 12 MB swizzled weights

    int totalFrag = 6 * NT * KCH * 32;
    prep_weights<<<(totalFrag + 255) / 256, 256, 0, stream>>>(
        enc_w_hh0, enc_w_ih1, enc_w_hh1, dec_w_hh0, dec_w_ih1, dec_w_hh1, frag);

    seq2seq_kernel<<<BB / 16, 512, 0, stream>>>(
        src, enc_w_ih0, enc_b0, enc_b1, dec_w_ih0, dec_b0, dec_b1,
        proj_w, proj_b, frag, (float*)d_out);
}
